// MPNNLSTM_20229295964650
// MI455X (gfx1250) — matrix operand order repaired and checked by reading, not compiled, not run on hardware
//
#include <hip/hip_runtime.h>

// ---------------------------------------------------------------------------
// MPNN-LSTM forward for MI455X (gfx1250, wave32, WMMA f16 16x16x32 / f32 acc)
// ---------------------------------------------------------------------------

#define WINDOW     7
#define NUM_NODES  20000
#define IN_CH      8
#define HID        64
#define N_TOTAL    (WINDOW * NUM_NODES)     // 140000
#define N_EDGES    (N_TOTAL * 16)           // 2240000
#define BN_EPS     1e-5f
#define OUT_COLS   (2 * HID + IN_CH + WINDOW - 1)   // 142

typedef __attribute__((ext_vector_type(16))) _Float16 v16h;
typedef __attribute__((ext_vector_type(8)))  _Float16 v8h;
typedef __attribute__((ext_vector_type(8)))  float    v8f;

// ---------------------------------------------------------------------------
// Fragment loaders (wave32); A and B both use contiguous K order k = 16*(lane/16) + s.
// Base pointers are lane-resolved outside the K loops; with compile-time
// strides every load below is base + immediate offset.
// ---------------------------------------------------------------------------

// A-matrix 16x32 f16 from a lane-resolved row pointer:
// lanes 0-15 hold K {0..15}; lanes 16-31 hold K {16..31}
// (the +16 half offset for hi lanes is folded into the base pointer).
__device__ __forceinline__ v16h make_A(const _Float16* p) {
    v8h lo = *(const v8h*)(p);
    v8h hh = *(const v8h*)(p + 8);
    v16h out;
#pragma unroll
    for (int i = 0; i < 8; ++i) { out[i] = lo[i]; out[i + 8] = hh[i]; }
    return out;
}

__device__ __forceinline__ float sigmoidf_(float x) {
    return 1.0f / (1.0f + __expf(-x));
}

// ---------------------------------------------------------------------------
// Small utility kernels
// ---------------------------------------------------------------------------

__global__ void fill_f32(float* __restrict__ p, float v, int n) {
    int i = blockIdx.x * blockDim.x + threadIdx.x;
    int s = gridDim.x * blockDim.x;
    for (; i < n; i += s) p[i] = v;
}

__global__ void fill_f16(_Float16* __restrict__ p, int n) {
    int i = blockIdx.x * blockDim.x + threadIdx.x;
    int s = gridDim.x * blockDim.x;
    for (; i < n; i += s) p[i] = (_Float16)0.0f;
}

// deg[dst] += w  (deg pre-filled with 1.0 for the self loop)
__global__ void deg_edges(const int* __restrict__ dst, const float* __restrict__ w,
                          float* __restrict__ deg, int E) {
    int e = blockIdx.x * blockDim.x + threadIdx.x;
    if (e >= E) return;
    __hip_atomic_fetch_add(&deg[dst[e]], w[e], __ATOMIC_RELAXED, __HIP_MEMORY_SCOPE_AGENT);
}

// deg -> dinv in place
__global__ void dinv_k(float* __restrict__ d, int n) {
    int i = blockIdx.x * blockDim.x + threadIdx.x;
    if (i >= n) return;
    float v = d[i];
    d[i] = (v > 0.0f) ? rsqrtf(v) : 0.0f;
}

__global__ void norm_k(const int* __restrict__ src, const int* __restrict__ dst,
                       const float* __restrict__ w, const float* __restrict__ dinv,
                       float* __restrict__ norm, int E) {
    int e = blockIdx.x * blockDim.x + threadIdx.x;
    if (e >= E) return;
    norm[e] = dinv[src[e]] * w[e] * dinv[dst[e]];
}

// Build f16 K-major weights for WMMA B operands + summed biases.
__global__ void prep_weights(const float* __restrict__ Wih1, const float* __restrict__ Whh1,
                             const float* __restrict__ bih1, const float* __restrict__ bhh1,
                             const float* __restrict__ Wih2, const float* __restrict__ Whh2,
                             const float* __restrict__ bih2, const float* __restrict__ bhh2,
                             const float* __restrict__ W2,
                             _Float16* __restrict__ Bcat1,   // [256][192]
                             _Float16* __restrict__ Bcat2,   // [256][128]
                             _Float16* __restrict__ W2T,     // [64][64]  W2T[n][k] = W2[k][n]
                             float* __restrict__ bsum1, float* __restrict__ bsum2) {
    int tid = blockIdx.x * blockDim.x + threadIdx.x;
    int str = gridDim.x * blockDim.x;
    for (int i = tid; i < 256 * 192; i += str) {
        int n = i / 192, k = i % 192;
        Bcat1[i] = (_Float16)(k < 128 ? Wih1[n * 128 + k] : Whh1[n * 64 + (k - 128)]);
    }
    for (int i = tid; i < 256 * 128; i += str) {
        int n = i / 128, k = i % 128;
        Bcat2[i] = (_Float16)(k < 64 ? Wih2[n * 64 + k] : Whh2[n * 64 + (k - 64)]);
    }
    for (int i = tid; i < 64 * 64; i += str) {
        int n = i / 64, k = i % 64;
        W2T[i] = (_Float16)W2[k * 64 + n];
    }
    for (int i = tid; i < 256; i += str) {
        bsum1[i] = bih1[i] + bhh1[i];
        bsum2[i] = bih2[i] + bhh2[i];
    }
}

// H = X @ W1   (K=8, memory bound -> plain VALU)
__global__ void gemm_in8(const float* __restrict__ X, const float* __restrict__ W1,
                         float* __restrict__ H) {
    int tid = blockIdx.x * blockDim.x + threadIdx.x;   // N_TOTAL*64 exact
    int row = tid >> 6, c = tid & 63;
    float acc = 0.0f;
#pragma unroll
    for (int k = 0; k < IN_CH; ++k) acc = fmaf(X[(size_t)row * IN_CH + k], W1[k * 64 + c], acc);
    H[tid] = acc;
}

// AGG = H * dinv^2 (self loop) + bias
__global__ void agg_init(const float* __restrict__ H, const float* __restrict__ dinv,
                         const float* __restrict__ b, float* __restrict__ AGG) {
    int tid = blockIdx.x * blockDim.x + threadIdx.x;   // N_TOTAL*64 exact
    int row = tid >> 6, c = tid & 63;
    float di = dinv[row];
    AGG[tid] = H[tid] * di * di + b[c];
}

// AGG[dst] += H[src] * norm ; 16 threads per edge, float4 per thread.
__global__ void scatter_edges(const int* __restrict__ src, const int* __restrict__ dst,
                              const float* __restrict__ norm, const float* __restrict__ H,
                              float* __restrict__ AGG) {
    long long gid = (long long)blockIdx.x * blockDim.x + threadIdx.x; // E*16 exact
    int e = (int)(gid >> 4);
    int q = ((int)gid & 15) << 2;
    float nr = norm[e];
    int s = src[e], d = dst[e];
    float4 m = *(const float4*)(H + (size_t)s * 64 + q);
    float* out = AGG + (size_t)d * 64 + q;
    __hip_atomic_fetch_add(out + 0, m.x * nr, __ATOMIC_RELAXED, __HIP_MEMORY_SCOPE_AGENT);
    __hip_atomic_fetch_add(out + 1, m.y * nr, __ATOMIC_RELAXED, __HIP_MEMORY_SCOPE_AGENT);
    __hip_atomic_fetch_add(out + 2, m.z * nr, __ATOMIC_RELAXED, __HIP_MEMORY_SCOPE_AGENT);
    __hip_atomic_fetch_add(out + 3, m.w * nr, __ATOMIC_RELAXED, __HIP_MEMORY_SCOPE_AGENT);
}

// In-place ReLU + per-channel sum / sumsq accumulation (LDS then global atomics).
__global__ void bn_pass1(float* __restrict__ AGG, float* __restrict__ gsum,
                         float* __restrict__ gsq, int total) {
    __shared__ float ls[128];
    if (threadIdx.x < 128) ls[threadIdx.x] = 0.0f;
    __syncthreads();
    int tid = blockIdx.x * blockDim.x + threadIdx.x;
    int str = gridDim.x * blockDim.x;                 // multiple of 64
    int c = tid & 63;
    float s = 0.0f, q = 0.0f;
    for (int i = tid; i < total; i += str) {
        float y = AGG[i];
        y = y > 0.0f ? y : 0.0f;
        AGG[i] = y;
        s += y; q += y * y;
    }
    __hip_atomic_fetch_add(&ls[c],      s, __ATOMIC_RELAXED, __HIP_MEMORY_SCOPE_WORKGROUP);
    __hip_atomic_fetch_add(&ls[64 + c], q, __ATOMIC_RELAXED, __HIP_MEMORY_SCOPE_WORKGROUP);
    __syncthreads();
    if (threadIdx.x < 64) {
        __hip_atomic_fetch_add(&gsum[threadIdx.x], ls[threadIdx.x],      __ATOMIC_RELAXED, __HIP_MEMORY_SCOPE_AGENT);
        __hip_atomic_fetch_add(&gsq[threadIdx.x],  ls[64 + threadIdx.x], __ATOMIC_RELAXED, __HIP_MEMORY_SCOPE_AGENT);
    }
}

__global__ void bn_finalize(const float* __restrict__ sum, const float* __restrict__ sq,
                            const float* __restrict__ gamma, const float* __restrict__ beta,
                            float* __restrict__ scale, float* __restrict__ shift) {
    int c = threadIdx.x;
    if (c < 64) {
        const float invN = 1.0f / (float)N_TOTAL;
        float m = sum[c] * invN;
        float v = sq[c] * invN - m * m;           // biased variance
        float s = gamma[c] * rsqrtf(v + BN_EPS);
        scale[c] = s;
        shift[c] = beta[c] - m * s;
    }
}

// Xcat[row][base+c] = f16( AGG * scale + shift )
__global__ void bn_apply(const float* __restrict__ AGG, const float* __restrict__ scale,
                         const float* __restrict__ shift, _Float16* __restrict__ Xcat, int base) {
    int tid = blockIdx.x * blockDim.x + threadIdx.x;   // N_TOTAL*64 exact
    int row = tid >> 6, c = tid & 63;
    Xcat[(size_t)row * 128 + base + c] = (_Float16)(AGG[tid] * scale[c] + shift[c]);
}

// ---------------------------------------------------------------------------
// WMMA GEMM: C[M x NT*16] = A[M x KT*32] * BT^T  (f16 in, f32 out).
// Strides are template constants -> base + immediate-offset addressing.
// B fragments of a k-tile are preloaded as a group so loads overlap WMMAs.
// ---------------------------------------------------------------------------
template <int NT, int KT, int LDA, int LDB, int LDC>
__global__ __launch_bounds__(256, 1) void wmma_gemm(
        const _Float16* __restrict__ A,
        const _Float16* __restrict__ BT,
        float* __restrict__ C, int M) {
    int wave = blockIdx.x * (blockDim.x >> 5) + (threadIdx.x >> 5);
    int m0 = wave << 4;
    if (m0 >= M) return;                 // wave-uniform: EXEC all-ones at WMMA
    int lane = threadIdx.x & 31;
    int r15 = lane & 15, hi = lane >> 4;

    const _Float16* a_base = A + (size_t)(m0 + r15) * LDA + hi * 16;
    const _Float16* b_base = BT + (size_t)r15 * LDB + (hi << 4);

    v8f acc[NT];
#pragma unroll
    for (int j = 0; j < NT; ++j) { v8f z = {}; acc[j] = z; }
#pragma unroll
    for (int kt = 0; kt < KT; ++kt) {
        v16h a = make_A(a_base + kt * 32);
        v16h bf[NT];
#pragma unroll
        for (int j = 0; j < NT; ++j)
            bf[j] = *(const v16h*)(b_base + (size_t)(j * 16) * LDB + kt * 32);
#pragma unroll
        for (int j = 0; j < NT; ++j)
            acc[j] = __builtin_amdgcn_wmma_f32_16x16x32_f16(false, a, false, bf[j],
                                                            (short)0, acc[j], false, false);
    }
    float* c_base = C + (size_t)(m0 + 8 * hi) * LDC + r15;
#pragma unroll
    for (int j = 0; j < NT; ++j)
#pragma unroll
        for (int r = 0; r < 8; ++r)
            c_base[(size_t)r * LDC + j * 16] = acc[j][r];
}

// ---------------------------------------------------------------------------
// Fused LSTM step. One wave = 16 rows x 16 columns of EACH gate:
// accumulator tiles {j0, j0+4, j0+8, j0+12} are exactly the i|f|g|o columns
// c = 16*j0 + (lane&15) -> cell fuses in registers with 4 acc fragments.
// The 4 waves of a row tile each read all 64 h cols of it but write only 16; no barrier orders this.
// ---------------------------------------------------------------------------
template <int NKT0, int NKT1, int LDA0, int LDA1, int LDB>
__global__ __launch_bounds__(256, 1) void lstm_step(
        const _Float16* __restrict__ A0,
        const _Float16* __restrict__ A1,
        const _Float16* __restrict__ BT,
        const float* __restrict__ bsum,
        float* __restrict__ cbuf,
        _Float16* __restrict__ h16,
        float* __restrict__ h32,
        int rows) {
    int wave = blockIdx.x * (blockDim.x >> 5) + (threadIdx.x >> 5);
    int j0 = wave & 3;                  // gate-column group (16 cols)
    int m0 = (wave >> 2) << 4;          // row tile
    if (m0 >= rows) return;             // wave-uniform
    int lane = threadIdx.x & 31;
    int r15 = lane & 15, hi = lane >> 4;

    const _Float16* a0_base = A0 + (size_t)(m0 + r15) * LDA0 + hi * 16;
    const _Float16* a1_base = A1 + (size_t)(m0 + r15) * LDA1 + hi * 16;
    // B rows for this wave: n = 16*(j0 + 4*g) + r15, g = gate index
    const _Float16* b_base = BT + (size_t)(j0 * 16 + r15) * LDB + (hi << 4);

    v8f acc[4];
#pragma unroll
    for (int g = 0; g < 4; ++g) { v8f z = {}; acc[g] = z; }

#pragma unroll
    for (int kt = 0; kt < NKT0; ++kt) {
        v16h a = make_A(a0_base + kt * 32);
        v16h bf[4];
#pragma unroll
        for (int g = 0; g < 4; ++g)
            bf[g] = *(const v16h*)(b_base + (size_t)(g * 64) * LDB + kt * 32);
#pragma unroll
        for (int g = 0; g < 4; ++g)
            acc[g] = __builtin_amdgcn_wmma_f32_16x16x32_f16(false, a, false, bf[g],
                                                            (short)0, acc[g], false, false);
    }
#pragma unroll
    for (int kt = 0; kt < NKT1; ++kt) {
        v16h a = make_A(a1_base + kt * 32);
        v16h bf[4];
#pragma unroll
        for (int g = 0; g < 4; ++g)
            bf[g] = *(const v16h*)(b_base + (size_t)(g * 64) * LDB + NKT0 * 32 + kt * 32);
#pragma unroll
        for (int g = 0; g < 4; ++g)
            acc[g] = __builtin_amdgcn_wmma_f32_16x16x32_f16(false, a, false, bf[g],
                                                            (short)0, acc[g], false, false);
    }

    // --- fused cell (PyTorch gate order i,f,g,o) ---
    int c = j0 * 16 + r15;
    float bi = bsum[c], bf_ = bsum[64 + c], bg = bsum[128 + c], bo = bsum[192 + c];
    size_t base = (size_t)(m0 + 8 * hi) * 64 + c;      // + r*64 per row
    float*    cb = cbuf + base;
    _Float16* hb = h16 + base;
    float*    wb = h32 + base;
#pragma unroll
    for (int r = 0; r < 8; ++r) {
        float ig = sigmoidf_(acc[0][r] + bi);
        float fg = sigmoidf_(acc[1][r] + bf_);
        float gg = tanhf(acc[2][r] + bg);
        float og = sigmoidf_(acc[3][r] + bo);
        float cn = fg * cb[r * 64] + ig * gg;
        float hn = og * tanhf(cn);
        cb[r * 64] = cn;
        hb[r * 64] = (_Float16)hn;
        wb[r * 64] = hn;
    }
}

// out[v] = [ h1 (64) | h2 (64) | x[v,0..7] | x[t*N+v,7] for t=1..6 ]
__global__ void assemble_out(const float* __restrict__ h1, const float* __restrict__ h2,
                             const float* __restrict__ x, float* __restrict__ out) {
    int tid = blockIdx.x * blockDim.x + threadIdx.x;
    if (tid >= NUM_NODES * OUT_COLS) return;
    int v = tid / OUT_COLS, j = tid % OUT_COLS;
    float val;
    if (j < 64)          val = h1[(size_t)v * 64 + j];
    else if (j < 128)    val = h2[(size_t)v * 64 + (j - 64)];
    else if (j < 136)    val = x[(size_t)v * IN_CH + (j - 128)];
    else {
        int t = j - 135;  // 1..6
        val = x[((size_t)t * NUM_NODES + v) * IN_CH + (IN_CH - 1)];
    }
    out[tid] = val;
}

// ---------------------------------------------------------------------------
// Orchestration
// ---------------------------------------------------------------------------
extern "C" void kernel_launch(void* const* d_in, const int* in_sizes, int n_in,
                              void* d_out, int out_size, void* d_ws, size_t ws_size,
                              hipStream_t stream) {
    (void)in_sizes; (void)n_in; (void)out_size; (void)ws_size;

    const float* x      = (const float*)d_in[0];
    const int*   ei     = (const int*)  d_in[1];   // [2][E]: src then dst
    const float* ew     = (const float*)d_in[2];
    const float* W1     = (const float*)d_in[3];
    const float* b1     = (const float*)d_in[4];
    const float* W2     = (const float*)d_in[5];
    const float* b2     = (const float*)d_in[6];
    const float* gamma1 = (const float*)d_in[7];
    const float* beta1  = (const float*)d_in[8];
    const float* gamma2 = (const float*)d_in[9];
    const float* beta2  = (const float*)d_in[10];
    const float* Wih1   = (const float*)d_in[11];
    const float* Whh1   = (const float*)d_in[12];
    const float* bih1   = (const float*)d_in[13];
    const float* bhh1   = (const float*)d_in[14];
    const float* Wih2   = (const float*)d_in[15];
    const float* Whh2   = (const float*)d_in[16];
    const float* bih2   = (const float*)d_in[17];
    const float* bhh2   = (const float*)d_in[18];

    const int* srcp = ei;
    const int* dstp = ei + N_EDGES;

    // ---- workspace carve (~143 MB) ----
    char* p = (char*)d_ws;
    auto carve = [&](size_t bytes) {
        void* r = (void*)p;
        p += (bytes + 255) & ~(size_t)255;
        return r;
    };
    float*    dinv   = (float*)   carve((size_t)N_TOTAL * 4);
    float*    normb  = (float*)   carve((size_t)N_EDGES * 4);
    float*    Hdense = (float*)   carve((size_t)N_TOTAL * 64 * 4);
    float*    AGG    = (float*)   carve((size_t)N_TOTAL * 64 * 4);
    _Float16* Xcat   = (_Float16*)carve((size_t)N_TOTAL * 128 * 2);
    _Float16* Bcat1  = (_Float16*)carve((size_t)256 * 192 * 2);
    _Float16* Bcat2  = (_Float16*)carve((size_t)256 * 128 * 2);
    _Float16* W2T    = (_Float16*)carve((size_t)64 * 64 * 2);
    float*    bsum1  = (float*)   carve(256 * 4);
    float*    bsum2  = (float*)   carve(256 * 4);
    float*    statb  = (float*)   carve(128 * 4);   // sum[64] | sq[64]
    float*    scaleb = (float*)   carve(64 * 4);
    float*    shiftb = (float*)   carve(64 * 4);
    float*    c1     = (float*)   carve((size_t)NUM_NODES * 64 * 4);
    float*    c2     = (float*)   carve((size_t)NUM_NODES * 64 * 4);
    _Float16* h1_16  = (_Float16*)carve((size_t)NUM_NODES * 64 * 2);
    _Float16* h2_16  = (_Float16*)carve((size_t)NUM_NODES * 64 * 2);
    float*    h1_32  = (float*)   carve((size_t)NUM_NODES * 64 * 4);
    float*    h2_32  = (float*)   carve((size_t)NUM_NODES * 64 * 4);

    const int EB   = (N_EDGES + 255) / 256;
    const int NF   = N_TOTAL * 64;         // 8,960,000
    const int NFB  = NF / 256;             // 35000 exact
    const int SCB  = (N_EDGES * 16) / 256; // 140000 exact

    // ---- GCN normalization ----
    fill_f32<<<(N_TOTAL + 255) / 256, 256, 0, stream>>>(dinv, 1.0f, N_TOTAL); // self-loop w=1
    deg_edges<<<EB, 256, 0, stream>>>(dstp, ew, dinv, N_EDGES);
    dinv_k<<<(N_TOTAL + 255) / 256, 256, 0, stream>>>(dinv, N_TOTAL);
    norm_k<<<EB, 256, 0, stream>>>(srcp, dstp, ew, dinv, normb, N_EDGES);
    prep_weights<<<128, 256, 0, stream>>>(Wih1, Whh1, bih1, bhh1, Wih2, Whh2, bih2, bhh2,
                                          W2, Bcat1, Bcat2, W2T, bsum1, bsum2);

    // ---- GCN layer 1: x @ W1 -> scatter -> ReLU -> BN -> Xcat[:,0:64] ----
    gemm_in8<<<NFB, 256, 0, stream>>>(x, W1, Hdense);
    agg_init<<<NFB, 256, 0, stream>>>(Hdense, dinv, b1, AGG);
    scatter_edges<<<SCB, 256, 0, stream>>>(srcp, dstp, normb, Hdense, AGG);
    fill_f32<<<1, 128, 0, stream>>>(statb, 0.0f, 128);
    bn_pass1<<<896, 256, 0, stream>>>(AGG, statb, statb + 64, NF);
    bn_finalize<<<1, 64, 0, stream>>>(statb, statb + 64, gamma1, beta1, scaleb, shiftb);
    bn_apply<<<NFB, 256, 0, stream>>>(AGG, scaleb, shiftb, Xcat, 0);

    // ---- GCN layer 2: X1 @ W2 (WMMA) -> scatter -> ReLU -> BN -> Xcat[:,64:128] ----
    wmma_gemm<4, 2, 128, 64, 64><<<(N_TOTAL / 16 + 7) / 8, 256, 0, stream>>>(Xcat, W2T,
                                                                             Hdense, N_TOTAL);
    agg_init<<<NFB, 256, 0, stream>>>(Hdense, dinv, b2, AGG);
    scatter_edges<<<SCB, 256, 0, stream>>>(srcp, dstp, normb, Hdense, AGG);
    fill_f32<<<1, 128, 0, stream>>>(statb, 0.0f, 128);
    bn_pass1<<<896, 256, 0, stream>>>(AGG, statb, statb + 64, NF);
    bn_finalize<<<1, 64, 0, stream>>>(statb, statb + 64, gamma2, beta2, scaleb, shiftb);
    bn_apply<<<NFB, 256, 0, stream>>>(AGG, scaleb, shiftb, Xcat, 64);

    // ---- stacked LSTM over T=7 (zero initial state) ----
    const int HN = NUM_NODES * 64;
    fill_f32<<<(HN + 255) / 256, 256, 0, stream>>>(c1, 0.0f, HN);
    fill_f32<<<(HN + 255) / 256, 256, 0, stream>>>(c2, 0.0f, HN);
    fill_f16<<<(HN + 255) / 256, 256, 0, stream>>>(h1_16, HN);
    fill_f16<<<(HN + 255) / 256, 256, 0, stream>>>(h2_16, HN);

    // 4 waves per 16-row tile -> (20000/16)*4 = 5000 waves = 625 blocks x 8 waves
    const int LSTM_BLOCKS = (NUM_NODES / 16) * 4 / 8;   // 625 exact
    for (int t = 0; t < WINDOW; ++t) {
        const _Float16* xt = Xcat + (size_t)t * NUM_NODES * 128;
        lstm_step<4, 2, 128, 64, 192><<<LSTM_BLOCKS, 256, 0, stream>>>(
            xt, h1_16, Bcat1, bsum1, c1, h1_16, h1_32, NUM_NODES);
        lstm_step<2, 2, 64, 64, 128><<<LSTM_BLOCKS, 256, 0, stream>>>(
            h1_16, h2_16, Bcat2, bsum2, c2, h2_16, h2_32, NUM_NODES);
    }

    // ---- output: [H1 | H2 | S] = [20000 x 142] ----
    assemble_out<<<(NUM_NODES * OUT_COLS + 255) / 256, 256, 0, stream>>>(h1_32, h2_32, x,
                                                                        (float*)d_out);
}
